// MultiHeadAttention_2808908612331
// MI455X (gfx1250) — matrix operand order repaired and checked by reading, not compiled, not run on hardware
//
#include <hip/hip_runtime.h>
#include <hip/hip_bf16.h>

typedef __attribute__((ext_vector_type(16))) _Float16 v16h;
typedef __attribute__((ext_vector_type(8)))  _Float16 v8h;
typedef __attribute__((ext_vector_type(8)))  float    v8f;
typedef __attribute__((ext_vector_type(4)))  float    v4f;
typedef __attribute__((ext_vector_type(4)))  int      v4i;

#define B_  2
#define L_  2048
#define D_  1024
#define H_  16
#define HD_ 64

#if defined(__has_builtin)
#if __has_builtin(__builtin_amdgcn_global_load_async_to_lds_b128)
#define HAS_ASYNC 1
#endif
#endif
#ifndef HAS_ASYNC
#define HAS_ASYNC 0
#endif

// A-fragment K pattern (16-bit A 16x32): VGPR j<4 -> K=khalf+2j, j>=4 -> K=khalf+16+2(j-4)
__device__ __forceinline__ int kpatA(int j, int khalf) {
    return khalf + ((j < 4) ? 2 * j : 16 + 2 * (j - 4));
}

template <int N>
__device__ __forceinline__ void wait_async() {
#if __has_builtin(__builtin_amdgcn_s_wait_asynccnt)
    __builtin_amdgcn_s_wait_asynccnt(N);
#else
    asm volatile("s_wait_asynccnt %0" ::"i"(N) : "memory");
#endif
    asm volatile("" ::: "memory");   // keep LDS reads below the wait
}

// Async-stage one 32(K)x64(N) f16 weight tile (4 KB) into LDS: 8 x b128 per lane.
__device__ __forceinline__ void stage_wtile(const _Float16* __restrict__ W,
                                            _Float16* lds,   // 4*16*32 halves
                                            int n0, int nc, int bkhalf, int k0)
{
#if HAS_ASYNC
#pragma unroll
    for (int nf = 0; nf < 4; ++nf) {
#pragma unroll
        for (int p = 0; p < 2; ++p) {
            const _Float16* g = W + (size_t)(n0 + nf * 16 + nc) * D_ + k0 + bkhalf + p * 8;
            _Float16* l = lds + (nf * 16 + nc) * 32 + bkhalf + p * 8;
            __builtin_amdgcn_global_load_async_to_lds_b128((v4i*)g, (v4i*)l, 0, 0);
        }
    }
#else
    (void)W; (void)lds; (void)n0; (void)nc; (void)bkhalf; (void)k0;
#endif
}

// ---------------------------------------------------------------------------
// Kernel 0: transpose + f16-convert weights: W f32 [in,out] -> Wt f16 [out,in]
// grid = (D/16, D/16, 4), block = (16,16)
// ---------------------------------------------------------------------------
__global__ void wt_kernel(const float* __restrict__ Wq,
                          const float* __restrict__ Wk,
                          const float* __restrict__ Wv,
                          const float* __restrict__ Wo,
                          _Float16* __restrict__ WtBase)
{
    __shared__ float tile[16][17];
    const int z = blockIdx.z;
    const float* W = (z == 0) ? Wq : (z == 1) ? Wk : (z == 2) ? Wv : Wo;
    _Float16* Wt = WtBase + (size_t)z * D_ * D_;

    const int tx = threadIdx.x, ty = threadIdx.y;
    const int k0 = blockIdx.x * 16;   // in-dim tile
    const int n0 = blockIdx.y * 16;   // out-dim tile

    tile[ty][tx] = W[(size_t)(k0 + ty) * D_ + n0 + tx];   // coalesced read
    __syncthreads();
    Wt[(size_t)(n0 + ty) * D_ + k0 + tx] = (_Float16)tile[tx][ty];  // coalesced write
}

// ---------------------------------------------------------------------------
// Kernel 1: fused QKV projection + bias + RoPE
//   q,k -> f16 [B,H,L,hd];  v -> f16 [B,H,hd,L] (transposed for P@V B-frags)
// grid = (D/64, (B*L)/16, 3), block = 32
// B tiles double-buffered through LDS via GLOBAL_LOAD_ASYNC_TO_LDS (ASYNCcnt)
// ---------------------------------------------------------------------------
__global__ void qkv_rope_kernel(const float* __restrict__ Q,
                                const float* __restrict__ K,
                                const float* __restrict__ V,
                                const _Float16* __restrict__ Wt,   // 4 x [D,D] f16
                                const float* __restrict__ bq,
                                const float* __restrict__ bk,
                                const float* __restrict__ bv,
                                _Float16* __restrict__ qh,
                                _Float16* __restrict__ kh,
                                _Float16* __restrict__ vt)
{
    const int z = blockIdx.z;
    const float*    X    = (z == 0) ? Q  : ((z == 1) ? K  : V);
    const _Float16* W    = Wt + (size_t)z * D_ * D_;
    const float*    bias = (z == 0) ? bq : ((z == 1) ? bk : bv);

    const int lane   = threadIdx.x;
    const int nc     = lane & 15;
    const int khalf  = (lane >> 4) << 3;   // 0 or 8  (A / C row offset)
    const int bkhalf = (lane >> 4) << 4;   // 0 or 16 (B K offset)
    const int m0     = blockIdx.y * 16;
    const int n0     = blockIdx.x * 64;
    const int arow   = m0 + nc;

    v8f acc[4] = {};

#if HAS_ASYNC
    __shared__ _Float16 lds_w[2][4 * 16 * 32];   // 2 x 4 KB double buffer
    stage_wtile(W, lds_w[0], n0, nc, bkhalf, 0);
#endif

    for (int k0 = 0; k0 < D_; k0 += 32) {
        // A: one contiguous run of 16 f32 -> 4x b128 loads, pack to f16
        const float* xa = X + (size_t)arow * D_ + k0 + bkhalf;
        const v4f r0 = *reinterpret_cast<const v4f*>(xa);
        const v4f r1 = *reinterpret_cast<const v4f*>(xa + 4);
        const v4f r2 = *reinterpret_cast<const v4f*>(xa + 8);
        const v4f r3 = *reinterpret_cast<const v4f*>(xa + 12);
        if (k0 + 32 < D_) __builtin_prefetch(xa + 32, 0, 0);
        v16h a;
#pragma unroll
        for (int i = 0; i < 4; ++i) {
            a[i]      = (_Float16)r0[i];
            a[4 + i]  = (_Float16)r1[i];
            a[8 + i]  = (_Float16)r2[i];
            a[12 + i] = (_Float16)r3[i];
        }
#if HAS_ASYNC
        const int cur = (k0 >> 5) & 1;
        if (k0 + 32 < D_) {
            stage_wtile(W, lds_w[cur ^ 1], n0, nc, bkhalf, k0 + 32); // prefetch next
            wait_async<8>();   // drain older group only (in-order completion)
        } else {
            wait_async<0>();
        }
#pragma unroll
        for (int nf = 0; nf < 4; ++nf) {
            const v16h bf = *reinterpret_cast<const v16h*>(
                &lds_w[cur][(nf * 16 + nc) * 32 + bkhalf]);
            acc[nf] = __builtin_amdgcn_wmma_f32_16x16x32_f16(
                false, a, false, bf, (short)0, acc[nf], false, false);
        }
#else
#pragma unroll
        for (int nf = 0; nf < 4; ++nf) {
            const int col = n0 + nf * 16 + nc;
            const v16h bf = *reinterpret_cast<const v16h*>(
                W + (size_t)col * D_ + k0 + bkhalf);
            acc[nf] = __builtin_amdgcn_wmma_f32_16x16x32_f16(
                false, a, false, bf, (short)0, acc[nf], false, false);
        }
#endif
    }

    // bias
#pragma unroll
    for (int nf = 0; nf < 4; ++nf) {
        const float bvv = bias[n0 + nf * 16 + nc];
#pragma unroll
        for (int r = 0; r < 8; ++r) acc[nf][r] += bvv;
    }

    // RoPE for q,k: pairs (d, d+32) are frags nf and nf+2 of the same lane
    if (z < 2) {
        const float inf0 = __powf(10000.0f, -((float)(nc))      / 32.0f);
        const float inf1 = __powf(10000.0f, -((float)(16 + nc)) / 32.0f);
#pragma unroll
        for (int r = 0; r < 8; ++r) {
            const int   gm = m0 + r + khalf;
            const float tf = (float)(gm & (L_ - 1));
            float s0, c0, s1, c1;
            __sincosf(tf * inf0, &s0, &c0);
            __sincosf(tf * inf1, &s1, &c1);
            const float x0 = acc[0][r], x1 = acc[1][r];
            const float x2 = acc[2][r], x3 = acc[3][r];
            acc[0][r] = x0 * c0 - x2 * s0;
            acc[2][r] = x2 * c0 + x0 * s0;
            acc[1][r] = x1 * c1 - x3 * s1;
            acc[3][r] = x3 * c1 + x1 * s1;
        }
    }

    const int bb = (m0 >> 11);          // batch (rows of a tile share b)
    const int t0 = m0 & (L_ - 1);
    if (z < 2) {
        // q/k row-major [B,H,L,hd]
        _Float16* Out = (z == 0) ? qh : kh;
#pragma unroll
        for (int r = 0; r < 8; ++r) {
            const int t = t0 + r + khalf;
#pragma unroll
            for (int nf = 0; nf < 4; ++nf) {
                const int col = n0 + nf * 16 + nc;
                const int h = col >> 6, d = col & 63;
                Out[(((size_t)(bb * H_ + h)) * L_ + t) * HD_ + d] = (_Float16)acc[nf][r];
            }
        }
    } else {
        // v transposed [B,H,hd,L]: 8 rows contiguous in t -> one v8h store per frag
#pragma unroll
        for (int nf = 0; nf < 4; ++nf) {
            const int col = n0 + nf * 16 + nc;
            const int h = col >> 6, d = col & 63;
            v8h vv;
#pragma unroll
            for (int r = 0; r < 8; ++r) vv[r] = (_Float16)acc[nf][r];
            *reinterpret_cast<v8h*>(
                vt + (((size_t)(bb * H_ + h)) * HD_ + d) * L_ + t0 + khalf) = vv;
        }
    }
}

// ---------------------------------------------------------------------------
// Kernel 2: causal flash attention, one wave per (b, h, 16-query tile)
// grid = (L/16, H, B), block = 32
// ---------------------------------------------------------------------------
__global__ void attn_kernel(const _Float16* __restrict__ qh,
                            const _Float16* __restrict__ kh,
                            const _Float16* __restrict__ vt,
                            _Float16* __restrict__ oh)
{
    __shared__ _Float16 lds_p[32 * 16];   // column-major: [col][row]

    const int lane   = threadIdx.x;
    const int nc     = lane & 15;
    const int khalf  = (lane >> 4) << 3;
    const int bkhalf = (lane >> 4) << 4;
    const int qt = blockIdx.x, h = blockIdx.y, b = blockIdx.z;
    const int q0 = qt * 16;
    const size_t headOff  = ((size_t)(b * H_ + h)) * L_ * HD_;   // q/k row-major
    const size_t headOffT = ((size_t)(b * H_ + h)) * HD_ * L_;   // v transposed

    // Q A-fragments: two 16x32 chunks over hd (2x v8h each)
    v16h aq[2];
#pragma unroll
    for (int c = 0; c < 2; ++c) {
        const _Float16* qp = qh + headOff + (size_t)(q0 + nc) * HD_ + c * 32 + bkhalf;
        const v8h lo = *reinterpret_cast<const v8h*>(qp);
        const v8h hi = *reinterpret_cast<const v8h*>(qp + 8);
#pragma unroll
        for (int i = 0; i < 8; ++i) { aq[c][i] = lo[i]; aq[c][8 + i] = hi[i]; }
    }

    v8f o[4] = {};
    float mrow[8], lrow[8];
#pragma unroll
    for (int r = 0; r < 8; ++r) { mrow[r] = -3.0e38f; lrow[r] = 0.0f; }

    const int nch = qt / 2 + 1;           // 32-key chunks up to the diagonal
    for (int j2 = 0; j2 < nch; ++j2) {
        const int kb = j2 * 32;

        // S = Q @ K^T for two 16x16 key tiles (B-frags contiguous 32B)
        v8f s[2] = {};
#pragma unroll
        for (int t = 0; t < 2; ++t) {
            const int key = kb + t * 16 + nc;
#pragma unroll
            for (int c = 0; c < 2; ++c) {
                const v16h bk = *reinterpret_cast<const v16h*>(
                    kh + headOff + (size_t)key * HD_ + c * 32 + bkhalf);
                s[t] = __builtin_amdgcn_wmma_f32_16x16x32_f16(
                    false, aq[c], false, bk, (short)0, s[t], false, false);
            }
        }

        // scale + causal mask (no mask tensor read)
#pragma unroll
        for (int t = 0; t < 2; ++t) {
            const int key = kb + t * 16 + nc;
#pragma unroll
            for (int r = 0; r < 8; ++r) {
                const int qrow = q0 + r + khalf;
                const float v = s[t][r] * 0.125f;     // 1/sqrt(64)
                s[t][r] = (key > qrow) ? -1.0e9f : v;
            }
        }

        // online softmax: 16-lane butterfly -> row stats land in C-layout slots
        v8h pw0, pw1;
#pragma unroll
        for (int r = 0; r < 8; ++r) {
            float mx = fmaxf(s[0][r], s[1][r]);
#pragma unroll
            for (int mk = 1; mk < 16; mk <<= 1) mx = fmaxf(mx, __shfl_xor(mx, mk, 32));
            const float mnew = fmaxf(mrow[r], mx);
            const float corr = __expf(mrow[r] - mnew);
            const float e0 = __expf(s[0][r] - mnew);
            const float e1 = __expf(s[1][r] - mnew);
            float sum = e0 + e1;
#pragma unroll
            for (int mk = 1; mk < 16; mk <<= 1) sum += __shfl_xor(sum, mk, 32);
            lrow[r] = lrow[r] * corr + sum;
            mrow[r] = mnew;
#pragma unroll
            for (int n = 0; n < 4; ++n) o[n][r] *= corr;
            pw0[r] = (_Float16)e0;
            pw1[r] = (_Float16)e1;
        }

        // P -> LDS column-major: one 16B ds store per tile per lane
        *reinterpret_cast<v8h*>(&lds_p[(nc)      * 16 + khalf]) = pw0;
        *reinterpret_cast<v8h*>(&lds_p[(16 + nc) * 16 + khalf]) = pw1;
        __syncthreads();

        // A-layout readback (row = nc, K runs along columns)
        v16h ap;
#pragma unroll
        for (int j = 0; j < 8; ++j) {
            const int kk = bkhalf + 2 * j;
            ap[2 * j]     = lds_p[kk * 16 + nc];
            ap[2 * j + 1] = lds_p[(kk + 1) * 16 + nc];
        }
        __syncthreads();

        // O += P @ V : B-frags contiguous 32B from transposed V
#pragma unroll
        for (int n = 0; n < 4; ++n) {
            const int d = n * 16 + nc;
            const v16h bv = *reinterpret_cast<const v16h*>(
                vt + headOffT + (size_t)d * L_ + kb + bkhalf);
            o[n] = __builtin_amdgcn_wmma_f32_16x16x32_f16(
                false, ap, false, bv, (short)0, o[n], false, false);
        }
    }

    // normalize, store f16 [B, L, D]
#pragma unroll
    for (int r = 0; r < 8; ++r) {
        const float inv = 1.0f / lrow[r];
        const int qrow = q0 + r + khalf;
        const size_t ob = ((size_t)(b * L_) + qrow) * D_ + h * HD_;
#pragma unroll
        for (int n = 0; n < 4; ++n)
            oh[ob + n * 16 + nc] = (_Float16)(o[n][r] * inv);
    }
}

// ---------------------------------------------------------------------------
// Kernel 3: output projection O @ Wo^T(f16) + bo, f32 out
// grid = (D/64, (B*L)/16), block = 32; async double-buffered B tiles
// ---------------------------------------------------------------------------
__global__ void oproj_kernel(const _Float16* __restrict__ oh,
                             const _Float16* __restrict__ Wot,  // [out,in] f16
                             const float* __restrict__ bo,
                             float* __restrict__ out)
{
    const int lane   = threadIdx.x;
    const int nc     = lane & 15;
    const int khalf  = (lane >> 4) << 3;
    const int bkhalf = (lane >> 4) << 4;
    const int m0 = blockIdx.y * 16;
    const int n0 = blockIdx.x * 64;
    const int arow = m0 + nc;

    v8f acc[4] = {};

#if HAS_ASYNC
    __shared__ _Float16 lds_w[2][4 * 16 * 32];
    stage_wtile(Wot, lds_w[0], n0, nc, bkhalf, 0);
#endif

    for (int k0 = 0; k0 < D_; k0 += 32) {
        const _Float16* oa = oh + (size_t)arow * D_ + k0 + bkhalf;
        const v8h lo = *reinterpret_cast<const v8h*>(oa);
        const v8h hi = *reinterpret_cast<const v8h*>(oa + 8);
        if (k0 + 32 < D_) __builtin_prefetch(oa + 32, 0, 0);
        v16h a;
#pragma unroll
        for (int i = 0; i < 8; ++i) { a[i] = lo[i]; a[8 + i] = hi[i]; }
#if HAS_ASYNC
        const int cur = (k0 >> 5) & 1;
        if (k0 + 32 < D_) {
            stage_wtile(Wot, lds_w[cur ^ 1], n0, nc, bkhalf, k0 + 32);
            wait_async<8>();
        } else {
            wait_async<0>();
        }
#pragma unroll
        for (int nf = 0; nf < 4; ++nf) {
            const v16h bf = *reinterpret_cast<const v16h*>(
                &lds_w[cur][(nf * 16 + nc) * 32 + bkhalf]);
            acc[nf] = __builtin_amdgcn_wmma_f32_16x16x32_f16(
                false, a, false, bf, (short)0, acc[nf], false, false);
        }
#else
#pragma unroll
        for (int nf = 0; nf < 4; ++nf) {
            const int col = n0 + nf * 16 + nc;
            const v16h bf = *reinterpret_cast<const v16h*>(
                Wot + (size_t)col * D_ + k0 + bkhalf);
            acc[nf] = __builtin_amdgcn_wmma_f32_16x16x32_f16(
                false, a, false, bf, (short)0, acc[nf], false, false);
        }
#endif
    }

#pragma unroll
    for (int r = 0; r < 8; ++r) {
        const int gm = m0 + r + khalf;
#pragma unroll
        for (int nf = 0; nf < 4; ++nf) {
            const int col = n0 + nf * 16 + nc;
            out[(size_t)gm * D_ + col] = acc[nf][r] + bo[col];
        }
    }
}

// ---------------------------------------------------------------------------
extern "C" void kernel_launch(void* const* d_in, const int* in_sizes, int n_in,
                              void* d_out, int out_size, void* d_ws, size_t ws_size,
                              hipStream_t stream)
{
    (void)in_sizes; (void)n_in; (void)out_size; (void)ws_size;

    const float* Q  = (const float*)d_in[0];
    const float* K  = (const float*)d_in[1];
    const float* V  = (const float*)d_in[2];
    const float* Wq = (const float*)d_in[3];
    const float* Wk = (const float*)d_in[4];
    const float* Wv = (const float*)d_in[5];
    const float* Wo = (const float*)d_in[6];
    const float* bq = (const float*)d_in[7];
    const float* bk = (const float*)d_in[8];
    const float* bv = (const float*)d_in[9];
    const float* bo = (const float*)d_in[10];
    // d_in[11] = causal mask: never read (causality applied analytically)

    char* ws = (char*)d_ws;
    const size_t seg  = (size_t)B_ * H_ * L_ * HD_ * sizeof(_Float16);  // 8 MiB
    const size_t wseg = (size_t)D_ * D_ * sizeof(_Float16);             // 2 MiB
    _Float16* qh  = (_Float16*)(ws);
    _Float16* kh  = (_Float16*)(ws + seg);
    _Float16* vtp = (_Float16*)(ws + 2 * seg);
    _Float16* oh  = (_Float16*)(ws + 3 * seg);
    _Float16* Wt  = (_Float16*)(ws + 4 * seg);   // 4 x 2 MiB: Wq,Wk,Wv,Wo transposed

    wt_kernel<<<dim3(D_ / 16, D_ / 16, 4), dim3(16, 16), 0, stream>>>(
        Wq, Wk, Wv, Wo, Wt);

    qkv_rope_kernel<<<dim3(D_ / 64, (B_ * L_) / 16, 3), 32, 0, stream>>>(
        Q, K, V, Wt, bq, bk, bv, qh, kh, vtp);

    attn_kernel<<<dim3(L_ / 16, H_, B_), 32, 0, stream>>>(qh, kh, vtp, oh);

    oproj_kernel<<<dim3(D_ / 64, (B_ * L_) / 16), 32, 0, stream>>>(
        oh, Wt + 3 * (wseg / sizeof(_Float16)), bo, (float*)d_out);
}
